// CrossAttentionP_52286931862006
// MI455X (gfx1250) — matrix operand order repaired and checked by reading, not compiled, not run on hardware
//
#include <hip/hip_runtime.h>

// Cross-attention with spatial reduction (PVT SRA), gfx1250 WMMA implementation.
// B=8, N=4096, C=512, heads=8, d=64, SR=2 -> M=1024 keys, H=W=64.

#define DIMC   512
#define HEADS  8
#define HDIM   64
#define BATCH  8
#define NQ     4096
#define MKV    1024
#define CC     (DIMC * DIMC)
#define LN_EPS 1e-5f

typedef __attribute__((ext_vector_type(16))) _Float16 v16h;
typedef __attribute__((ext_vector_type(8)))  _Float16 v8h;
typedef __attribute__((ext_vector_type(8)))  float    v8f;

// ---------------------------------------------------------------------------
// WMMA helpers (CDNA5: V_WMMA_F32_16X16X32_F16, wave32)
// ---------------------------------------------------------------------------
static __device__ __forceinline__ v8f wmma32(v16h a, v16h b, v8f c) {
  // 8 args: (neg_a, A, neg_b, B, c_mod, C, reuse_a, reuse_b)
  return __builtin_amdgcn_wmma_f32_16x16x32_f16(false, a, false, b, (short)0, c,
                                                false, false);
}

static __device__ __forceinline__ v16h frag_join(v8h lo, v8h hi) {
  v16h r;
#pragma unroll
  for (int i = 0; i < 8; ++i) { r[i] = lo[i]; r[i + 8] = hi[i]; }
  return r;
}

// A fragment: 16(M) x 32(K) f16 from row-major src (ld in elements).
// Lane L: row = L&15, k-offsets {8h..8h+7} and {16+8h..16+8h+7}, h = L>>4.
static __device__ __forceinline__ v16h load_a(const _Float16* __restrict__ src,
                                              int ld) {
  const int lane = threadIdx.x & 31;
  const _Float16* p = src + (size_t)(lane & 15) * ld + ((lane >> 4) << 3);
  return frag_join(*(const v8h*)p, *(const v8h*)(p + 16));
}

// B fragment: 32(K) x 16(N) f16 from COLUMN-major B == row-major [n][k] src.
// Lane L: row n = L&15, k-offsets {8h..8h+7} and {16+8h..16+8h+7}, h = L>>4.
static __device__ __forceinline__ v16h load_b(const _Float16* __restrict__ src,
                                              int ld) {
  const int lane = threadIdx.x & 31;
  const _Float16* p = src + (size_t)(lane & 15) * ld + ((lane >> 4) << 3);
  return frag_join(*(const v8h*)p, *(const v8h*)(p + 16));
}

// Shuffle reductions over 16-lane halves (keeps D-fragment row halves apart).
static __device__ __forceinline__ float rmax16(float x) {
#pragma unroll
  for (int m = 1; m < 16; m <<= 1) x = fmaxf(x, __shfl_xor(x, m, 32));
  return x;
}
static __device__ __forceinline__ float rsum16(float x) {
#pragma unroll
  for (int m = 1; m < 16; m <<= 1) x += __shfl_xor(x, m, 32);
  return x;
}

// ---------------------------------------------------------------------------
// Elementwise conversion kernels
// ---------------------------------------------------------------------------
__global__ void cvt_f32_f16_kernel(const float* __restrict__ src,
                                   _Float16* __restrict__ dst, int n) {
  for (int i = blockIdx.x * blockDim.x + threadIdx.x; i < n;
       i += gridDim.x * blockDim.x)
    dst[i] = (_Float16)src[i];
}

// Wsr is OIHW [C,C,2,2]; emit 4 per-tap f16 matrices [tap][out][in].
__global__ void cvt_wsr_kernel(const float* __restrict__ src,
                               _Float16* __restrict__ dst) {
  const int n = 4 * CC;
  for (int idx = blockIdx.x * blockDim.x + threadIdx.x; idx < n;
       idx += gridDim.x * blockDim.x) {
    int s = idx >> 18;            // tap = kh*2+kw
    int rem = idx & (CC - 1);     // o*C + i
    dst[idx] = (_Float16)src[((size_t)rem << 2) + s];
  }
}

// ---------------------------------------------------------------------------
// Generic GEMM: out[m,n] = sum_k A[m,k] * W[n,k] + bias[n]
// A: f16 [Mrows x 512], W: f16 [512 x 512] (torch [out,in] layout),
// wave tile = 32(M) x 64(N). EP: 0 = f32 store, 1 = f16 store,
// 2 = f16 transposed-per-batch store into [B, C, 1024] (for V^T).
// ---------------------------------------------------------------------------
template <int EP>
__global__ __launch_bounds__(256) void gemm_kernel(
    const _Float16* __restrict__ A, const _Float16* __restrict__ W,
    const float* __restrict__ bias, void* __restrict__ outp, int Mrows) {
  const int wave = blockIdx.x * (blockDim.x >> 5) + (threadIdx.x >> 5);
  const int tm = wave >> 3, tn = wave & 7;  // 8 n-tiles of 64 over N=512
  if (tm * 32 >= Mrows) return;
  const int lane = threadIdx.x & 31;

  v8f acc[2][4] = {};
  const _Float16* a0p = A + (size_t)tm * 32 * DIMC;
  const _Float16* a1p = a0p + (size_t)16 * DIMC;
  for (int k = 0; k < DIMC; k += 32) {
    v16h a0 = load_a(a0p + k, DIMC);
    v16h a1 = load_a(a1p + k, DIMC);
#pragma unroll
    for (int j = 0; j < 4; ++j) {
      v16h b = load_b(W + (size_t)(tn * 64 + j * 16) * DIMC + k, DIMC);
      acc[0][j] = wmma32(a0, b, acc[0][j]);
      acc[1][j] = wmma32(a1, b, acc[1][j]);
    }
  }

  const int n0 = lane & 15, mh8 = (lane >> 4) << 3;
#pragma unroll
  for (int j = 0; j < 4; ++j) {
    const int nglob = tn * 64 + j * 16 + n0;
    const float bv = bias[nglob];
#pragma unroll
    for (int rh = 0; rh < 2; ++rh) {
#pragma unroll
      for (int i = 0; i < 8; ++i) {
        const int m = tm * 32 + rh * 16 + mh8 + i;
        const float val = acc[rh][j][i] + bv;
        if (EP == 0) {
          ((float*)outp)[(size_t)m * DIMC + nglob] = val;
        } else if (EP == 1) {
          ((_Float16*)outp)[(size_t)m * DIMC + nglob] = (_Float16)val;
        } else {  // transposed within batch: [b][c][key]
          const int bb = m >> 10, key = m & 1023;
          ((_Float16*)outp)[((size_t)bb * DIMC + nglob) * MKV + key] =
              (_Float16)val;
        }
      }
    }
  }
}

// ---------------------------------------------------------------------------
// SR conv (k=s=2) as 4 accumulated implicit GEMMs.
// out f32 [B*1024, 512]; A rows gathered from context rows (2mh+kh)*64+2mw+kw.
// ---------------------------------------------------------------------------
__global__ __launch_bounds__(256) void conv_sr_kernel(
    const _Float16* __restrict__ ctxh, const _Float16* __restrict__ Wsr,
    const float* __restrict__ bias, float* __restrict__ out) {
  const int wave = blockIdx.x * (blockDim.x >> 5) + (threadIdx.x >> 5);
  const int tm = wave >> 3, tn = wave & 7;
  const int lane = threadIdx.x & 31;
  const int r = lane & 15, hh = lane >> 4;

  const int mAll0 = tm * 32 + r, mAll1 = mAll0 + 16;
  const int b0 = mAll0 >> 10, m0 = mAll0 & 1023;
  const int b1 = mAll1 >> 10, m1 = mAll1 & 1023;
  const int mh0 = m0 >> 5, mw0 = m0 & 31;
  const int mh1 = m1 >> 5, mw1 = m1 & 31;

  v8f acc[2][4] = {};
#pragma unroll
  for (int s = 0; s < 4; ++s) {
    const int kh = s >> 1, kw = s & 1;
    const _Float16* w = Wsr + (size_t)s * CC;
    const _Float16* ar0 =
        ctxh + ((size_t)b0 * NQ + (2 * mh0 + kh) * 64 + 2 * mw0 + kw) * DIMC +
        (hh << 3);
    const _Float16* ar1 =
        ctxh + ((size_t)b1 * NQ + (2 * mh1 + kh) * 64 + 2 * mw1 + kw) * DIMC +
        (hh << 3);
    for (int k = 0; k < DIMC; k += 32) {
      v16h a0 = frag_join(*(const v8h*)(ar0 + k), *(const v8h*)(ar0 + k + 16));
      v16h a1 = frag_join(*(const v8h*)(ar1 + k), *(const v8h*)(ar1 + k + 16));
#pragma unroll
      for (int j = 0; j < 4; ++j) {
        v16h b = load_b(w + (size_t)(tn * 64 + j * 16) * DIMC + k, DIMC);
        acc[0][j] = wmma32(a0, b, acc[0][j]);
        acc[1][j] = wmma32(a1, b, acc[1][j]);
      }
    }
  }

  const int n0 = lane & 15, mh8 = (lane >> 4) << 3;
#pragma unroll
  for (int j = 0; j < 4; ++j) {
    const int nglob = tn * 64 + j * 16 + n0;
    const float bv = bias[nglob];
#pragma unroll
    for (int rh = 0; rh < 2; ++rh)
#pragma unroll
      for (int i = 0; i < 8; ++i) {
        const int m = tm * 32 + rh * 16 + mh8 + i;
        out[(size_t)m * DIMC + nglob] = acc[rh][j][i] + bv;
      }
  }
}

// ---------------------------------------------------------------------------
// LayerNorm: one wave per row of 512, f32 in -> f16 out.
// ---------------------------------------------------------------------------
__global__ __launch_bounds__(256) void ln_kernel(
    const float* __restrict__ x, const float* __restrict__ g,
    const float* __restrict__ be, _Float16* __restrict__ out, int rows) {
  const int wave = blockIdx.x * (blockDim.x >> 5) + (threadIdx.x >> 5);
  if (wave >= rows) return;
  const int lane = threadIdx.x & 31;
  const float* row = x + (size_t)wave * DIMC;
  float vals[16], s = 0.f, s2 = 0.f;
#pragma unroll
  for (int i = 0; i < 16; ++i) {
    float v = row[lane + i * 32];
    vals[i] = v; s += v; s2 += v * v;
  }
#pragma unroll
  for (int m = 1; m < 32; m <<= 1) {
    s += __shfl_xor(s, m, 32);
    s2 += __shfl_xor(s2, m, 32);
  }
  const float mu = s * (1.f / DIMC);
  const float rstd = rsqrtf(s2 * (1.f / DIMC) - mu * mu + LN_EPS);
#pragma unroll
  for (int i = 0; i < 16; ++i) {
    const int c = lane + i * 32;
    out[(size_t)wave * DIMC + c] = (_Float16)((vals[i] - mu) * rstd * g[c] + be[c]);
  }
}

// ---------------------------------------------------------------------------
// Fused flash attention: one wave per (b, head, 16-query tile).
// qh: f16 [B*N, C]; kh: f16 [B*M, C]; vT: f16 [B, C, M]; ao: f16 [B*N, C].
// ---------------------------------------------------------------------------
__global__ __launch_bounds__(256) void attn_kernel(
    const _Float16* __restrict__ qh, const _Float16* __restrict__ khb,
    const _Float16* __restrict__ vT, _Float16* __restrict__ ao) {
  __shared__ _Float16 pbuf[8][16 * 32];  // per-wave P staging (f16, row-major)
  const int wid = threadIdx.x >> 5;
  const int wave = blockIdx.x * 8 + wid;
  const int qt = wave & 255;
  const int h = (wave >> 8) & 7;
  const int b = wave >> 11;
  const int lane = threadIdx.x & 31;
  const int n0 = lane & 15, mh8 = (lane >> 4) << 3;

  const _Float16* qb = qh + ((size_t)(b * NQ + qt * 16)) * DIMC + h * HDIM;
  const v16h aq0 = load_a(qb, DIMC);
  const v16h aq1 = load_a(qb + 32, DIMC);

  const _Float16* kb = khb + (size_t)b * MKV * DIMC + h * HDIM;
  const _Float16* vb = vT + ((size_t)b * DIMC + h * HDIM) * MKV;

  v8f o[4] = {};
  float mrow[8], lsum[8];
#pragma unroll
  for (int i = 0; i < 8; ++i) { mrow[i] = -3.0e38f; lsum[i] = 0.f; }

  _Float16* pb = pbuf[wid];
  const float scale = 0.125f;  // d^-0.5, d=64

  for (int j = 0; j < 32; ++j) {  // 32 keys per iteration, M=1024
    const _Float16* kp = kb + (size_t)(j * 32) * DIMC;
    if (j < 31) {
      __builtin_prefetch(kp + (size_t)32 * DIMC, 0, 3);        // next K chunk
      __builtin_prefetch(vb + (j + 1) * 32, 0, 3);             // next V chunk
    }
    // S tile: 16 q x 32 keys via 4 WMMAs
    v8f z = {};
    v16h b00 = load_b(kp, DIMC);
    v16h b01 = load_b(kp + 32, DIMC);
    v16h b10 = load_b(kp + (size_t)16 * DIMC, DIMC);
    v16h b11 = load_b(kp + (size_t)16 * DIMC + 32, DIMC);
    v8f s0 = wmma32(aq0, b00, z); s0 = wmma32(aq1, b01, s0);
    v8f s1 = wmma32(aq0, b10, z); s1 = wmma32(aq1, b11, s1);

    // Online softmax; D-fragment row mapping == O accumulator mapping.
#pragma unroll
    for (int i = 0; i < 8; ++i) {
      const float x0 = s0[i] * scale, x1 = s1[i] * scale;
      const float cm = rmax16(fmaxf(x0, x1));
      const float mn = fmaxf(mrow[i], cm);
      const float corr = __expf(mrow[i] - mn);
      const float p0 = __expf(x0 - mn), p1 = __expf(x1 - mn);
      lsum[i] = lsum[i] * corr + rsum16(p0 + p1);
      mrow[i] = mn;
      o[0][i] *= corr; o[1][i] *= corr; o[2][i] *= corr; o[3][i] *= corr;
      pb[(mh8 + i) * 32 + n0] = (_Float16)p0;
      pb[(mh8 + i) * 32 + 16 + n0] = (_Float16)p1;
    }
    // P f32(D-layout) -> f16 A-fragment via LDS round trip (same wave;
    // compiler inserts s_wait_dscnt for the RAW through shared memory).
    v16h ap = load_a(pb, 32);

    const _Float16* vp = vb + j * 32;
#pragma unroll
    for (int t = 0; t < 4; ++t) {
      v16h bv = load_b(vp + (size_t)(t * 16) * MKV, MKV);
      o[t] = wmma32(ap, bv, o[t]);
    }
  }

#pragma unroll
  for (int i = 0; i < 8; ++i) {
    const float inv = 1.f / lsum[i];
    o[0][i] *= inv; o[1][i] *= inv; o[2][i] *= inv; o[3][i] *= inv;
  }
  _Float16* ob = ao + ((size_t)(b * NQ + qt * 16)) * DIMC + h * HDIM;
#pragma unroll
  for (int t = 0; t < 4; ++t)
#pragma unroll
    for (int i = 0; i < 8; ++i)
      ob[(size_t)(mh8 + i) * DIMC + t * 16 + n0] = (_Float16)o[t][i];
}

// ---------------------------------------------------------------------------
// Host orchestration
// ---------------------------------------------------------------------------
extern "C" void kernel_launch(void* const* d_in, const int* in_sizes, int n_in,
                              void* d_out, int out_size, void* d_ws,
                              size_t ws_size, hipStream_t stream) {
  (void)in_sizes; (void)n_in; (void)out_size; (void)ws_size;
  const float* x    = (const float*)d_in[0];
  const float* ctx  = (const float*)d_in[1];
  const float* Wq   = (const float*)d_in[2];
  const float* bq   = (const float*)d_in[3];
  const float* Wk   = (const float*)d_in[4];
  const float* bk   = (const float*)d_in[5];
  const float* Wv   = (const float*)d_in[6];
  const float* bv   = (const float*)d_in[7];
  const float* Wp   = (const float*)d_in[8];
  const float* bp   = (const float*)d_in[9];
  const float* Wsrk = (const float*)d_in[10];
  const float* bsrk = (const float*)d_in[11];
  const float* Wsrv = (const float*)d_in[12];
  const float* bsrv = (const float*)d_in[13];
  const float* gk   = (const float*)d_in[14];
  const float* bek  = (const float*)d_in[15];
  const float* gv   = (const float*)d_in[16];
  const float* bev  = (const float*)d_in[17];

  char* ws = (char*)d_ws;
  size_t off = 0;
  auto alloc = [&](size_t bytes) -> char* {
    char* p = ws + off;
    off += (bytes + 255) & ~(size_t)255;
    return p;
  };

  _Float16* wqh  = (_Float16*)alloc((size_t)CC * 2);
  _Float16* wkh  = (_Float16*)alloc((size_t)CC * 2);
  _Float16* wvh  = (_Float16*)alloc((size_t)CC * 2);
  _Float16* wph  = (_Float16*)alloc((size_t)CC * 2);
  _Float16* wsrk = (_Float16*)alloc((size_t)4 * CC * 2);
  _Float16* wsrv = (_Float16*)alloc((size_t)4 * CC * 2);
  _Float16* xh   = (_Float16*)alloc((size_t)BATCH * NQ * DIMC * 2);
  _Float16* ctxh = (_Float16*)alloc((size_t)BATCH * NQ * DIMC * 2);
  float*    srk  = (float*)alloc((size_t)BATCH * MKV * DIMC * 4);
  float*    srv  = (float*)alloc((size_t)BATCH * MKV * DIMC * 4);
  _Float16* ckh  = (_Float16*)alloc((size_t)BATCH * MKV * DIMC * 2);
  _Float16* cvh  = (_Float16*)alloc((size_t)BATCH * MKV * DIMC * 2);
  _Float16* qhb  = (_Float16*)alloc((size_t)BATCH * NQ * DIMC * 2);
  _Float16* khb  = (_Float16*)alloc((size_t)BATCH * MKV * DIMC * 2);
  _Float16* vtb  = (_Float16*)alloc((size_t)BATCH * DIMC * MKV * 2);
  _Float16* aob  = (_Float16*)alloc((size_t)BATCH * NQ * DIMC * 2);

  // 1) f32 -> f16 conversions
  cvt_f32_f16_kernel<<<4096, 256, 0, stream>>>(x, xh, BATCH * NQ * DIMC);
  cvt_f32_f16_kernel<<<4096, 256, 0, stream>>>(ctx, ctxh, BATCH * NQ * DIMC);
  cvt_f32_f16_kernel<<<512, 256, 0, stream>>>(Wq, wqh, CC);
  cvt_f32_f16_kernel<<<512, 256, 0, stream>>>(Wk, wkh, CC);
  cvt_f32_f16_kernel<<<512, 256, 0, stream>>>(Wv, wvh, CC);
  cvt_f32_f16_kernel<<<512, 256, 0, stream>>>(Wp, wph, CC);
  cvt_wsr_kernel<<<1024, 256, 0, stream>>>(Wsrk, wsrk);
  cvt_wsr_kernel<<<1024, 256, 0, stream>>>(Wsrv, wsrv);

  // 2) spatial-reduction convs as implicit GEMMs (f32 out for LN)
  conv_sr_kernel<<<256, 256, 0, stream>>>(ctxh, wsrk, bsrk, srk);
  conv_sr_kernel<<<256, 256, 0, stream>>>(ctxh, wsrv, bsrv, srv);

  // 3) LayerNorm -> f16
  ln_kernel<<<1024, 256, 0, stream>>>(srk, gk, bek, ckh, BATCH * MKV);
  ln_kernel<<<1024, 256, 0, stream>>>(srv, gv, bev, cvh, BATCH * MKV);

  // 4) projections: q (f16), k (f16), v (f16 transposed -> [B,C,M])
  gemm_kernel<1><<<1024, 256, 0, stream>>>(xh, wqh, bq, qhb, BATCH * NQ);
  gemm_kernel<1><<<256, 256, 0, stream>>>(ckh, wkh, bk, khb, BATCH * MKV);
  gemm_kernel<2><<<256, 256, 0, stream>>>(cvh, wvh, bv, vtb, BATCH * MKV);

  // 5) fused flash attention (16384 waves)
  attn_kernel<<<2048, 256, 0, stream>>>(qhb, khb, vtb, aob);

  // 6) output projection -> f32 d_out
  gemm_kernel<0><<<1024, 256, 0, stream>>>(aob, wph, bp, (float*)d_out,
                                           BATCH * NQ);
}
